// FCOS_60215441489940
// MI455X (gfx1250) — compile-verified, matrix operand order checked by reading, not run on hardware
//
#include <hip/hip_runtime.h>
#include <hip/hip_bf16.h>
#include <math.h>

typedef __attribute__((ext_vector_type(16))) _Float16 v16h;
typedef __attribute__((ext_vector_type(8)))  _Float16 v8h;
typedef __attribute__((ext_vector_type(8)))  float    v8f;

#define NUM_CLASSES 20
#define NTOT_C      5376      // 64*64 + 32*32 + 16*16
#define KPRE        1000
#define IMG_SIZE_F  512.0f
#define SCORE_TH    0.3f
#define IOU_TH      0.5f

#define SB_STRIDE   40        // halfs per B-tile row (80B: 16B-aligned runs, conflict-free)

// -------------------------------------------------------------------------
// BN precompute: scale = gamma * rsqrt(var+eps); shift = beta - mean*scale
// -------------------------------------------------------------------------
__global__ void k_bn_pre(const float* __restrict__ gamma, const float* __restrict__ beta,
                         const float* __restrict__ mean,  const float* __restrict__ var,
                         float* __restrict__ scale, float* __restrict__ shift, int C) {
    int c = blockIdx.x * blockDim.x + threadIdx.x;
    if (c < C) {
        float s = gamma[c] * rsqrtf(var[c] + 1e-5f);
        scale[c] = s;
        shift[c] = beta[c] - mean[c] * s;
    }
}

// -------------------------------------------------------------------------
// Weight pre-transform: fp32 OIHW [Cout][Cin][ks][ks] -> f16 [Cout][KK][Cin]
// (window-position-major K order so conv staging is channel-contiguous).
// -------------------------------------------------------------------------
__global__ void k_wtrans(const float* __restrict__ in, _Float16* __restrict__ out,
                         int Cout, int Cin, int KK) {
    int idx = blockIdx.x * blockDim.x + threadIdx.x;
    int tot = Cout * KK * Cin;
    if (idx >= tot) return;
    int o   = idx / (KK * Cin);
    int rem = idx - o * KK * Cin;
    int r   = rem / Cin;
    int ci  = rem - r * Cin;
    out[idx] = (_Float16)in[((size_t)o * Cin + ci) * KK + r];
}

// -------------------------------------------------------------------------
// Implicit-GEMM conv, K order = r*Cin + ci (window-position-major).
// WMMA f32 <- f16 x f16 (16x16x32). Block = 256 threads = 8 waves.
// Block tile: 128 out-channels x 64 pixels; wave w holds couts [w*16,w*16+16)
// with 4 accumulators over 4 pixel sub-tiles -> 4 WMMA per K-step from one
// A fragment. B tile (32K x 64N) staged in LDS, double-buffered (1 barrier
// per K-step). Weights pre-transposed to f16 [Cout][KK][Cin].
// -------------------------------------------------------------------------
template <int KS>
__global__ __launch_bounds__(256)
void k_conv_wmma(const float* __restrict__ X, const _Float16* __restrict__ Wt2,
                 const float* __restrict__ bias, const float* __restrict__ bscale,
                 const float* __restrict__ bshift, int relu,
                 float* __restrict__ Y,
                 int B, int Cin, int Cout, int H, int W) {
    constexpr int KK  = KS * KS;
    constexpr int PAD = KS / 2;
    const int HW  = H * W;
    const int nci = (Cin + 31) >> 5;     // K-steps per window position
    const int S   = KK * nci;            // total K-steps

    const int b     = blockIdx.z;
    const int ntile = blockIdx.x * 64;
    const int wave  = threadIdx.x >> 5;
    const int lane  = threadIdx.x & 31;
    const int l     = lane & 15;
    const int hi    = lane >> 4;
    const int co_b  = blockIdx.y * 128 + wave * 16;

    __shared__ _Float16 sB[2][64 * SB_STRIDE];   // double-buffered B stage

    const float* xb = X + (size_t)b * Cin * HW;

    // ---- per-thread staging slot (fixed pixel, 8 channel-slices/step) ----
    const int  n_s  = threadIdx.x & 63;
    const int  kkb  = threadIdx.x >> 6;          // 0..3
    const int  ng   = ntile + n_s;
    const bool ngok = ng < HW;
    const int  py   = ng / W;
    const int  px   = ng - py * W;

    const int o = co_b + l;                       // A-fragment out-channel row

    // Stage K-step `step` into `buf`: B(ci_local, n) = x[ci0+ci_local][py+dy][px+dx]
    auto stage = [&](int step, _Float16* buf) {
        const int rr  = step / nci;
        const int ci0 = (step - rr * nci) << 5;
        const int dy  = rr / KS - PAD;
        const int dx  = rr - (rr / KS) * KS - PAD;
        const int sy  = py + dy, sx = px + dx;
        const bool okr = ngok && (unsigned)sy < (unsigned)H && (unsigned)sx < (unsigned)W;
        const int soff = sy * W + sx;             // valid only when okr
        #pragma unroll
        for (int t = 0; t < 8; ++t) {
            const int ci = ci0 + kkb + t * 4;
            float v = 0.0f;
            if (okr && ci < Cin) v = xb[(size_t)ci * HW + soff];
            buf[n_s * SB_STRIDE + kkb + t * 4] = (_Float16)v;
        }
    };

    v8f acc0 = {}, acc1 = {}, acc2 = {}, acc3 = {};

    stage(0, sB[0]);
    __syncthreads();

    for (int s = 0; s < S; ++s) {
        _Float16* cur = sB[s & 1];
        if (s + 1 < S) stage(s + 1, sB[(s + 1) & 1]);

        // ---- A fragment from pre-transposed f16 weights (two 16B runs) ----
        const int rr  = s / nci;
        const int ci0 = (s - rr * nci) << 5;
        const _Float16* wr = Wt2 + ((size_t)o * KK + rr) * Cin + ci0;
        v16h af;
        if (o < Cout && ci0 + 31 < Cin) {
            const v8h a0 = *(const v8h*)(wr + hi * 8);
            const v8h a1 = *(const v8h*)(wr + 16 + hi * 8);
            af = __builtin_shufflevector(a0, a1,
                     0, 1, 2, 3, 4, 5, 6, 7, 8, 9, 10, 11, 12, 13, 14, 15);
        } else {
            #pragma unroll
            for (int j = 0; j < 8; ++j) {
                const int c1 = ci0 + hi * 8 + j;
                af[j]     = (o < Cout && c1 < Cin)      ? wr[hi * 8 + j]      : (_Float16)0.0f;
                af[j + 8] = (o < Cout && c1 + 16 < Cin) ? wr[16 + hi * 8 + j] : (_Float16)0.0f;
            }
        }
        if (o < Cout && s + 1 < S) __builtin_prefetch(wr + 32, 0, 3);

        // ---- 4 x (B fragment + WMMA) ----
        #pragma unroll
        for (int nb = 0; nb < 4; ++nb) {
            const _Float16* brow = &cur[(nb * 16 + l) * SB_STRIDE];
            const v8h b0 = *(const v8h*)(brow + hi * 8);
            const v8h b1 = *(const v8h*)(brow + 16 + hi * 8);
            const v16h bf = __builtin_shufflevector(b0, b1,
                              0, 1, 2, 3, 4, 5, 6, 7, 8, 9, 10, 11, 12, 13, 14, 15);
            v8f* acc = (nb == 0) ? &acc0 : (nb == 1) ? &acc1 : (nb == 2) ? &acc2 : &acc3;
            *acc = __builtin_amdgcn_wmma_f32_16x16x32_f16(
                       false, af, false, bf, (short)0, *acc, false, false);
        }
        __syncthreads();
    }

    // ---- epilogue: C/D layout: VGPR r -> M = r + hi*8, N = l ----
    #pragma unroll
    for (int nb = 0; nb < 4; ++nb) {
        const v8f& acc = (nb == 0) ? acc0 : (nb == 1) ? acc1 : (nb == 2) ? acc2 : acc3;
        const int n = ntile + nb * 16 + l;
        #pragma unroll
        for (int r = 0; r < 8; ++r) {
            const int co = co_b + r + hi * 8;
            if (co < Cout && n < HW) {
                float v = acc[r];
                if (bias)   v += bias[co];
                if (bscale) v  = v * bscale[co] + bshift[co];
                if (relu)   v  = v > 0.0f ? v : 0.0f;
                Y[((size_t)b * Cout + co) * HW + n] = v;
            }
        }
    }
}

// -------------------------------------------------------------------------
// dst(H,W) += nearest-2x-upsample(small(H/2,W/2)), in place on dst.
// -------------------------------------------------------------------------
__global__ void k_up_add(float* __restrict__ dst, const float* __restrict__ sml,
                         int B, int C, int H, int W) {
    int idx = blockIdx.x * blockDim.x + threadIdx.x;
    int tot = B * C * H * W;
    if (idx >= tot) return;
    int x  = idx % W;
    int y  = (idx / W) % H;
    int bc = idx / (W * H);
    int Hs = H >> 1, Ws = W >> 1;
    dst[idx] += sml[((size_t)bc * Hs + (y >> 1)) * Ws + (x >> 1)];
}

// -------------------------------------------------------------------------
// Per-level decode: scores = sqrt(sig(cls)*sig(ctr)); max/argmax over class;
// box decode with clamp to [0, 512]. Appends into flat per-image arrays.
// -------------------------------------------------------------------------
__global__ void k_decode(const float* __restrict__ cls, const float* __restrict__ boxp,
                         const float* __restrict__ ctr,
                         float* __restrict__ boxesA, float* __restrict__ scoresA,
                         int* __restrict__ classesA,
                         int B, int H, int W, float stride, int loff) {
    int idx = blockIdx.x * blockDim.x + threadIdx.x;
    int HW  = H * W;
    if (idx >= B * HW) return;
    int b = idx / HW, n = idx - b * HW;

    float sct  = 1.0f / (1.0f + __expf(-ctr[(size_t)b * HW + n]));
    float best = -1.0f; int bc = 0;
    for (int c = 0; c < NUM_CLASSES; ++c) {
        float s  = 1.0f / (1.0f + __expf(-cls[((size_t)b * NUM_CLASSES + c) * HW + n]));
        float sc = sqrtf(s * sct);
        if (sc > best) { best = sc; bc = c; }
    }

    int py = n / W, px = n - (n / W) * W;
    float lx = ((float)px + 0.5f) * stride;
    float ly = ((float)py + 0.5f) * stride;
    float d[4];
    #pragma unroll
    for (int q = 0; q < 4; ++q) {
        float dv = boxp[((size_t)b * 4 + q) * HW + n] * stride;
        d[q] = dv > 0.0f ? dv : 0.0f;
    }
    float bx[4] = { lx - d[0], ly - d[1], lx + d[2], ly + d[3] };
    size_t o = (size_t)b * NTOT_C + loff + n;
    #pragma unroll
    for (int q = 0; q < 4; ++q)
        boxesA[o * 4 + q] = fminf(fmaxf(bx[q], 0.0f), IMG_SIZE_F);
    scoresA[o]  = best;
    classesA[o] = bc;
}

// -------------------------------------------------------------------------
// Exact top-K by rank counting (stable on ties -> matches lax.top_k).
// Score table is brought into LDS with gfx1250 async-to-LDS loads
// (ASYNCcnt-tracked), then masked in place.
// NOTE: sS is the only LDS allocation in this kernel -> LDS offset of
// sS[i] is i*4 bytes.
// -------------------------------------------------------------------------
__global__ __launch_bounds__(256)
void k_topk(const float* __restrict__ boxesA, const float* __restrict__ scoresA,
            const int* __restrict__ classesA,
            float* __restrict__ topB, float* __restrict__ topS, int* __restrict__ topC) {
    int b = blockIdx.x;
    __shared__ float sS[NTOT_C];

    // async gather: LDS[i*4] = scoresA[b*NTOT + i]
    for (int i = threadIdx.x; i < NTOT_C; i += 256) {
        unsigned lds_off = (unsigned)(i * 4);
        unsigned long long ga =
            (unsigned long long)(uintptr_t)(scoresA + (size_t)b * NTOT_C + i);
        asm volatile("global_load_async_to_lds_b32 %0, %1, off"
                     :: "v"(lds_off), "v"(ga) : "memory");
    }
    asm volatile("s_wait_asynccnt 0" ::: "memory");
    __syncthreads();

    // mask in place: masked = s > TH ? s : -1
    for (int i = threadIdx.x; i < NTOT_C; i += 256) {
        float s = sS[i];
        sS[i] = s > SCORE_TH ? s : -1.0f;
    }
    __syncthreads();

    for (int i = threadIdx.x; i < NTOT_C; i += 256) {
        float v = sS[i];
        int rank = 0;
        for (int m = 0; m < NTOT_C; ++m) {
            float u = sS[m];
            rank += (u > v) || (u == v && m < i);
        }
        if (rank < KPRE) {
            size_t src = (size_t)b * NTOT_C + i;
            size_t dst = (size_t)b * KPRE + rank;
            topS[dst] = scoresA[src];
            topC[dst] = classesA[src];
            #pragma unroll
            for (int q = 0; q < 4; ++q) topB[dst * 4 + q] = boxesA[src * 4 + q];
        }
    }
}

// -------------------------------------------------------------------------
// Sequential class-offset NMS per image + final masked output write.
// -------------------------------------------------------------------------
__global__ __launch_bounds__(256)
void k_nms(const float* __restrict__ topB, const float* __restrict__ topS,
           const int* __restrict__ topC, float* __restrict__ out) {
    int b   = blockIdx.x;
    int tid = threadIdx.x;
    __shared__ float bx[KPRE][4];
    __shared__ float off[KPRE];
    __shared__ float area[KPRE];
    __shared__ float sc[KPRE];
    __shared__ int   keep[KPRE];
    __shared__ float red[256];

    float lmax = -1e30f;
    for (int i = tid; i < KPRE; i += 256) {
        #pragma unroll
        for (int q = 0; q < 4; ++q) {
            float v = topB[((size_t)b * KPRE + i) * 4 + q];
            bx[i][q] = v;
            lmax = fmaxf(lmax, v);
        }
        float s = topS[(size_t)b * KPRE + i];
        sc[i]   = s;
        keep[i] = s > SCORE_TH ? 1 : 0;
    }
    red[tid] = lmax;
    __syncthreads();
    for (int s = 128; s > 0; s >>= 1) {
        if (tid < s) red[tid] = fmaxf(red[tid], red[tid + s]);
        __syncthreads();
    }
    float mx = red[0] + 1.0f;
    __syncthreads();

    for (int i = tid; i < KPRE; i += 256) {
        off[i] = (float)topC[(size_t)b * KPRE + i] * mx;
        float w = fmaxf(bx[i][2] - bx[i][0], 0.0f);
        float h = fmaxf(bx[i][3] - bx[i][1], 0.0f);
        area[i] = w * h;
    }
    __syncthreads();

    for (int i = 0; i < KPRE; ++i) {
        if (keep[i]) {
            float x1i = bx[i][0] + off[i], y1i = bx[i][1] + off[i];
            float x2i = bx[i][2] + off[i], y2i = bx[i][3] + off[i];
            float ai  = area[i];
            for (int j = i + 1 + tid; j < KPRE; j += 256) {
                if (keep[j]) {
                    float oj  = off[j];
                    float xx1 = fmaxf(x1i, bx[j][0] + oj);
                    float yy1 = fmaxf(y1i, bx[j][1] + oj);
                    float xx2 = fminf(x2i, bx[j][2] + oj);
                    float yy2 = fminf(y2i, bx[j][3] + oj);
                    float inter = fmaxf(xx2 - xx1, 0.0f) * fmaxf(yy2 - yy1, 0.0f);
                    float iou = inter / (ai + area[j] - inter + 1e-12f);
                    if (iou > IOU_TH) keep[j] = 0;
                }
            }
        }
        __syncthreads();
    }

    for (int j = tid; j < KPRE; j += 256) {
        float k = keep[j] ? 1.0f : 0.0f;
        size_t o = ((size_t)b * KPRE + j) * 5;
        #pragma unroll
        for (int q = 0; q < 4; ++q) out[o + q] = bx[j][q] * k;
        out[o + 4] = sc[j] * k;
    }
}

// -------------------------------------------------------------------------
// Host-side orchestration
// -------------------------------------------------------------------------
static void launch_conv(hipStream_t s, const float* x, const _Float16* w2,
                        const float* bias, const float* bscale, const float* bshift,
                        int relu, float* y,
                        int B, int Cin, int Cout, int H, int W, int ks) {
    dim3 grid((H * W + 63) / 64, (Cout + 127) / 128, B);
    if (ks == 1)
        k_conv_wmma<1><<<grid, 256, 0, s>>>(x, w2, bias, bscale, bshift, relu, y,
                                            B, Cin, Cout, H, W);
    else
        k_conv_wmma<3><<<grid, 256, 0, s>>>(x, w2, bias, bscale, bshift, relu, y,
                                            B, Cin, Cout, H, W);
}

extern "C" void kernel_launch(void* const* d_in, const int* in_sizes, int n_in,
                              void* d_out, int out_size, void* d_ws, size_t ws_size,
                              hipStream_t stream) {
    (void)in_sizes; (void)n_in; (void)out_size; (void)ws_size;
    const int B = 8, C = 128;

    // ---- inputs (jax pytree order: dict keys sorted, lists in order) ----
    const float* c3        = (const float*)d_in[0];   // (8,64,64,64)
    const float* c4        = (const float*)d_in[1];   // (8,160,32,32)
    const float* c5        = (const float*)d_in[2];   // (8,400,16,16)
    const float* bn_beta   = (const float*)d_in[3];
    const float* bn_gamma  = (const float*)d_in[4];
    const float* bn_mean   = (const float*)d_in[5];
    const float* bn_var    = (const float*)d_in[6];
    const float* lat_c3    = (const float*)d_in[7];   // (128,64,1,1)
    const float* lat_c4    = (const float*)d_in[8];   // (128,160,1,1)
    const float* lat_c5    = (const float*)d_in[9];   // (128,400,1,1)
    const float* o_p3_b1_w = (const float*)d_in[10];  // (128,128,3,3)
    const float* o_p3_b2_b = (const float*)d_in[11];
    const float* o_p3_b2_w = (const float*)d_in[12];
    const float* o_p4_b    = (const float*)d_in[13];
    const float* o_p4_w    = (const float*)d_in[14];
    const float* o_p5_b    = (const float*)d_in[15];
    const float* o_p5_w    = (const float*)d_in[16];
    const float* pred_box_b = (const float*)d_in[17];
    const float* pred_box_w = (const float*)d_in[18]; // (4,128,3,3)
    const float* pred_cls_b = (const float*)d_in[19];
    const float* pred_cls_w = (const float*)d_in[20]; // (20,128,3,3)
    const float* pred_ctr_b = (const float*)d_in[21];
    const float* pred_ctr_w = (const float*)d_in[22]; // (1,128,3,3)
    const float* stem_box_b0 = (const float*)d_in[23];
    const float* stem_box_b1 = (const float*)d_in[24];
    const float* stem_box_w0 = (const float*)d_in[25];
    const float* stem_box_w1 = (const float*)d_in[26];
    const float* stem_cls_b0 = (const float*)d_in[27];
    const float* stem_cls_b1 = (const float*)d_in[28];
    const float* stem_cls_w0 = (const float*)d_in[29];
    const float* stem_cls_w1 = (const float*)d_in[30];

    float* out = (float*)d_out;

    // ---- workspace layout ----
    float* ws = (float*)d_ws;
    size_t off = 0;
    auto alloc = [&](size_t n) -> float* {
        float* p = ws + off;
        off += (n + 63) & ~(size_t)63;
        return p;
    };
    auto alloc_h = [&](size_t n) -> _Float16* {     // n halfs
        return (_Float16*)alloc((n + 1) / 2);
    };
    float* l5   = alloc((size_t)B * C * 16 * 16);
    float* l4   = alloc((size_t)B * C * 32 * 32);
    float* l3   = alloc((size_t)B * C * 64 * 64);   // l3 -> m3 -> p3
    float* p5   = alloc((size_t)B * C * 16 * 16);
    float* p4   = alloc((size_t)B * C * 32 * 32);
    float* tbuf = alloc((size_t)B * C * 64 * 64);
    float* s1   = alloc((size_t)B * C * 64 * 64);
    float* s2   = alloc((size_t)B * C * 64 * 64);
    float* clsb = alloc((size_t)B * NUM_CLASSES * 64 * 64);
    float* boxb = alloc((size_t)B * 4 * 64 * 64);
    float* ctrb = alloc((size_t)B * 64 * 64);
    float* boxesA  = alloc((size_t)B * NTOT_C * 4);
    float* scoresA = alloc((size_t)B * NTOT_C);
    int*   classesA = (int*)alloc((size_t)B * NTOT_C);
    float* topB = alloc((size_t)B * KPRE * 4);
    float* topS = alloc((size_t)B * KPRE);
    int*   topC = (int*)alloc((size_t)B * KPRE);
    float* bnsc = alloc(C);
    float* bnsh = alloc(C);

    // f16 pre-transposed weights [Cout][KK][Cin]
    struct WT { const float* src; int Cout, Cin, KK; _Float16* dst; };
    WT wt[15] = {
        { lat_c3,      C,  64, 1, nullptr }, { lat_c4,      C, 160, 1, nullptr },
        { lat_c5,      C, 400, 1, nullptr }, { o_p5_w,      C,   C, 9, nullptr },
        { o_p4_w,      C,   C, 9, nullptr }, { o_p3_b1_w,   C,   C, 9, nullptr },
        { o_p3_b2_w,   C,   C, 9, nullptr }, { stem_cls_w0, C,   C, 9, nullptr },
        { stem_cls_w1, C,   C, 9, nullptr }, { stem_box_w0, C,   C, 9, nullptr },
        { stem_box_w1, C,   C, 9, nullptr }, { pred_cls_w, NUM_CLASSES, C, 9, nullptr },
        { pred_box_w,  4,   C, 9, nullptr }, { pred_ctr_w,  1,   C, 9, nullptr },
        { nullptr, 0, 0, 0, nullptr }
    };
    for (int i = 0; i < 14; ++i) {
        size_t tot = (size_t)wt[i].Cout * wt[i].KK * wt[i].Cin;
        wt[i].dst = alloc_h(tot);
        k_wtrans<<<((int)tot + 255) / 256, 256, 0, stream>>>(
            wt[i].src, wt[i].dst, wt[i].Cout, wt[i].Cin, wt[i].KK);
    }
    _Float16 *w_lat3 = wt[0].dst, *w_lat4 = wt[1].dst, *w_lat5 = wt[2].dst;
    _Float16 *w_p5 = wt[3].dst, *w_p4 = wt[4].dst, *w_p3b1 = wt[5].dst, *w_p3b2 = wt[6].dst;
    _Float16 *w_sc0 = wt[7].dst, *w_sc1 = wt[8].dst, *w_sb0 = wt[9].dst, *w_sb1 = wt[10].dst;
    _Float16 *w_pcls = wt[11].dst, *w_pbox = wt[12].dst, *w_pctr = wt[13].dst;

    // ---- FPN ----
    k_bn_pre<<<1, 128, 0, stream>>>(bn_gamma, bn_beta, bn_mean, bn_var, bnsc, bnsh, C);

    launch_conv(stream, c5, w_lat5, nullptr, nullptr, nullptr, 0, l5, B, 400, C, 16, 16, 1);
    launch_conv(stream, c4, w_lat4, nullptr, nullptr, nullptr, 0, l4, B, 160, C, 32, 32, 1);
    launch_conv(stream, c3, w_lat3, nullptr, nullptr, nullptr, 0, l3, B,  64, C, 64, 64, 1);

    launch_conv(stream, l5, w_p5, o_p5_b, nullptr, nullptr, 0, p5, B, C, C, 16, 16, 3);

    { int tot = B * C * 32 * 32;
      k_up_add<<<(tot + 255) / 256, 256, 0, stream>>>(l4, l5, B, C, 32, 32); }  // m4 in l4
    launch_conv(stream, l4, w_p4, o_p4_b, nullptr, nullptr, 0, p4, B, C, C, 32, 32, 3);

    { int tot = B * C * 64 * 64;
      k_up_add<<<(tot + 255) / 256, 256, 0, stream>>>(l3, l4, B, C, 64, 64); }  // m3 in l3
    launch_conv(stream, l3, w_p3b1, nullptr, bnsc, bnsh, 1, tbuf, B, C, C, 64, 64, 3);
    launch_conv(stream, tbuf, w_p3b2, o_p3_b2_b, nullptr, nullptr, 0, l3, B, C, C, 64, 64, 3); // p3

    // ---- heads + decode per level ----
    struct Lvl { const float* f; int H, W; float stride; int loff; };
    Lvl lv[3] = { { l3, 64, 64,  8.0f,    0 },
                  { p4, 32, 32, 16.0f, 4096 },
                  { p5, 16, 16, 32.0f, 5120 } };
    for (int li = 0; li < 3; ++li) {
        const Lvl& L = lv[li];
        int HW = L.H * L.W;
        // cls branch
        launch_conv(stream, L.f, w_sc0, stem_cls_b0, nullptr, nullptr, 1, s1, B, C, C, L.H, L.W, 3);
        launch_conv(stream, s1,  w_sc1, stem_cls_b1, nullptr, nullptr, 1, s2, B, C, C, L.H, L.W, 3);
        launch_conv(stream, s2,  w_pcls, pred_cls_b, nullptr, nullptr, 0, clsb, B, C, NUM_CLASSES, L.H, L.W, 3);
        // box branch
        launch_conv(stream, L.f, w_sb0, stem_box_b0, nullptr, nullptr, 1, s1, B, C, C, L.H, L.W, 3);
        launch_conv(stream, s1,  w_sb1, stem_box_b1, nullptr, nullptr, 1, s2, B, C, C, L.H, L.W, 3);
        launch_conv(stream, s2,  w_pbox, pred_box_b, nullptr, nullptr, 0, boxb, B, C, 4, L.H, L.W, 3);
        launch_conv(stream, s2,  w_pctr, pred_ctr_b, nullptr, nullptr, 0, ctrb, B, C, 1, L.H, L.W, 3);
        // decode
        int tot = B * HW;
        k_decode<<<(tot + 255) / 256, 256, 0, stream>>>(clsb, boxb, ctrb,
                                                        boxesA, scoresA, classesA,
                                                        B, L.H, L.W, L.stride, L.loff);
    }

    // ---- top-k + NMS + output ----
    k_topk<<<B, 256, 0, stream>>>(boxesA, scoresA, classesA, topB, topS, topC);
    k_nms<<<B, 256, 0, stream>>>(topB, topS, topC, out);
}
